// AttentionHead_28432683499824
// MI455X (gfx1250) — matrix operand order repaired and checked by reading, not compiled, not run on hardware
//
#include <hip/hip_runtime.h>
#include <hip/hip_bf16.h>

#define HIDDEN 1024
#define DIMK   64
#define BATCH  4
#define SEQ    4096
#define NROWS  (BATCH * SEQ)

typedef __attribute__((ext_vector_type(16))) __bf16 v16bf;
typedef __attribute__((ext_vector_type(8)))  float  v8f;
typedef int v4i __attribute__((vector_size(16)));

union FragBF { unsigned int u[8]; v16bf v; };

#define ASG __attribute__((address_space(1)))
#define ASL __attribute__((address_space(3)))

#if defined(__has_builtin)
#if __has_builtin(__builtin_amdgcn_global_load_async_to_lds_b128)
#define ASYNC_LDS 1
#define ASYNC_B128(gp, lp) \
    __builtin_amdgcn_global_load_async_to_lds_b128((ASG v4i*)(gp), (ASL v4i*)(lp), 0, 0)
#if __has_builtin(__builtin_amdgcn_s_wait_asynccnt)
#define WAIT_ASYNC() __builtin_amdgcn_s_wait_asynccnt(0)
#else
#define WAIT_ASYNC() asm volatile("s_wait_asynccnt 0x0" ::: "memory")
#endif
#endif
#endif

// Native bf16 conversion (RNE) - lets the backend use v_cvt_pk_bf16_f32.
__device__ __forceinline__ unsigned short f32_to_bf16(float f) {
    union { __bf16 h; unsigned short u; } p;
    p.h = (__bf16)f;
    return p.u;
}
__device__ __forceinline__ unsigned int pack_bf16(float lo, float hi) {
    union { __bf16 h[2]; unsigned int u; } p;
    p.h[0] = (__bf16)lo;
    p.h[1] = (__bf16)hi;
    return p.u;
}

// ---------------------------------------------------------------------------
// Kernel 1: P = X[16384x1024] @ W[1024x64] + b  ->  bf16 output
// block = 256 threads (8 waves), 128 rows per block.
// X chunk staged as f32 via async global->LDS b128; W chunk staged transposed
// as packed bf16 dwords.
// ---------------------------------------------------------------------------
__global__ __launch_bounds__(256) void proj_kernel(const float* __restrict__ X,
                                                   const float* __restrict__ W,
                                                   const float* __restrict__ bias,
                                                   unsigned short* __restrict__ Outb) {
    __shared__ __attribute__((aligned(16))) float          Xf[128 * 32];  // 16 KB
    __shared__ __attribute__((aligned(16))) unsigned short Wt[64 * 32];   // 4 KB, [n][k] bf16

    const int t     = threadIdx.x;
    const int lane  = t & 31;
    const int w     = t >> 5;
    const int khalf = lane >> 4;
    const int nlo   = lane & 15;
    const int rowBase = blockIdx.x * 128;

    v8f acc[4];
    for (int nt = 0; nt < 4; ++nt)
        for (int i = 0; i < 8; ++i) acc[nt][i] = 0.0f;

    unsigned int* WtU = (unsigned int*)Wt;

    for (int kt = 0; kt < HIDDEN / 32; ++kt) {
        const int k0 = kt * 32;

        // ---- stage X chunk (128 rows x 32 f32) : 1024 x 16B segments ----
#ifdef ASYNC_LDS
#pragma unroll
        for (int i = 0; i < 4; ++i) {
            int e  = t + i * 256;            // 0..1023
            int r  = e >> 3;
            int sc = (e & 7) * 4;
            const float* gp = X + (long)(rowBase + r) * HIDDEN + k0 + sc;
            float*       lp = &Xf[r * 32 + sc];
            ASYNC_B128(gp, lp);
        }
#else
#pragma unroll
        for (int i = 0; i < 4; ++i) {
            int e  = t + i * 256;
            int r  = e >> 3;
            int sc = (e & 7) * 4;
            const float4 vv = *(const float4*)(X + (long)(rowBase + r) * HIDDEN + k0 + sc);
            *(float4*)&Xf[r * 32 + sc] = vv;
        }
#endif
        // ---- stage W chunk transposed: Wt[n][k], packed bf16 pairs ----
#pragma unroll
        for (int i = 0; i < 4; ++i) {
            int e  = t + i * 256;            // 0..1023 dwords; e = kk*64 + n
            int n  = e & 63;
            int kk = e >> 6;                 // K-pair index 0..15
            float w0 = W[(long)(k0 + 2 * kk) * DIMK + n];
            float w1 = W[(long)(k0 + 2 * kk + 1) * DIMK + n];
            WtU[n * 16 + kk] = pack_bf16(w0, w1);
        }
#ifdef ASYNC_LDS
        WAIT_ASYNC();
#endif
        __syncthreads();

        // A fragment: pack f32 pairs from LDS per wave32 A-layout
        FragBF a;
        const float* xr = &Xf[(w * 16 + nlo) * 32];
#pragma unroll
        for (int j = 0; j < 8; ++j) {
            int koff = khalf * 16 + 2 * j;
            a.u[j] = pack_bf16(xr[koff], xr[koff + 1]);
        }
#pragma unroll
        for (int nt = 0; nt < 4; ++nt) {
            FragBF bfr;
            const unsigned int* wr = WtU + (nt * 16 + nlo) * 16;
#pragma unroll
            for (int j = 0; j < 8; ++j) bfr.u[j] = wr[khalf * 8 + j];  // K = khalf*16 + 2j
            acc[nt] = __builtin_amdgcn_wmma_f32_16x16x32_bf16(
                false, a.v, false, bfr.v, (short)0, acc[nt], false, false);
        }
        __syncthreads();
    }

    // epilogue: +bias, convert to bf16, store
#pragma unroll
    for (int nt = 0; nt < 4; ++nt) {
        int n = nt * 16 + nlo;
        float bvv = bias[n];
#pragma unroll
        for (int r = 0; r < 8; ++r) {
            int row = rowBase + w * 16 + r + (khalf ? 8 : 0);
            Outb[(long)row * DIMK + n] = f32_to_bf16(acc[nt][r] + bvv);
        }
    }
}

// ---------------------------------------------------------------------------
// Kernel 2: scores = (Q @ K^T) * 0.125 per batch, 128x128 tiles, f32 out.
// Q/K tiles (16 KB contiguous bf16) staged with async global->LDS b128.
// ---------------------------------------------------------------------------
__global__ __launch_bounds__(256) void scores_kernel(const unsigned short* __restrict__ Qb,
                                                     const unsigned short* __restrict__ Kb,
                                                     float* __restrict__ Sc) {
    __shared__ __attribute__((aligned(16))) unsigned short Qs[128 * 64];  // 16 KB
    __shared__ __attribute__((aligned(16))) unsigned short Ks[128 * 64];  // 16 KB

    const int t     = threadIdx.x;
    const int lane  = t & 31;
    const int w     = t >> 5;
    const int khalf = lane >> 4;
    const int nlo   = lane & 15;
    const int qbase = blockIdx.x * 128;
    const int kbase = blockIdx.y * 128;
    const int b     = blockIdx.z;

    const unsigned int* Qg = (const unsigned int*)(Qb + (long)(b * SEQ + qbase) * DIMK);
    const unsigned int* Kg = (const unsigned int*)(Kb + (long)(b * SEQ + kbase) * DIMK);
    unsigned int* Qsu = (unsigned int*)Qs;
    unsigned int* Ksu = (unsigned int*)Ks;

#ifdef ASYNC_LDS
#pragma unroll
    for (int i = 0; i < 4; ++i) {
        int e = t + i * 256;                 // 1024 x 16B segments each
        ASYNC_B128(Qg + e * 4, Qsu + e * 4);
        ASYNC_B128(Kg + e * 4, Ksu + e * 4);
    }
    WAIT_ASYNC();
#else
    for (int i = 0; i < 16; ++i) {
        int e = t + i * 256;
        Qsu[e] = Qg[e];
        Ksu[e] = Kg[e];
    }
#endif
    __syncthreads();

    v8f acc[8];
    for (int nt = 0; nt < 8; ++nt)
        for (int i = 0; i < 8; ++i) acc[nt][i] = 0.0f;

#pragma unroll
    for (int kk = 0; kk < 2; ++kk) {             // Dk = 64 = 2 x 32
        FragBF a;
        const unsigned int* qr = Qsu + (w * 16 + nlo) * 32;
#pragma unroll
        for (int j = 0; j < 4; ++j) {
            a.u[j]     = qr[kk * 16 + khalf * 4 + j];
            a.u[j + 4] = qr[kk * 16 + 8 + khalf * 4 + j];
        }
#pragma unroll
        for (int nt = 0; nt < 8; ++nt) {
            FragBF bb;  // B[k][n] = K[kbase + nt*16 + n][k]  (K^T)
            const unsigned int* kr = Ksu + (nt * 16 + nlo) * 32;
#pragma unroll
            for (int j = 0; j < 8; ++j) bb.u[j] = kr[kk * 16 + khalf * 4 + (j & 3) + (j >> 2) * 8];
            acc[nt] = __builtin_amdgcn_wmma_f32_16x16x32_bf16(
                false, a.v, false, bb.v, (short)0, acc[nt], false, false);
        }
    }

#pragma unroll
    for (int nt = 0; nt < 8; ++nt)
#pragma unroll
        for (int r = 0; r < 8; ++r) {
            long row = (long)b * SEQ + qbase + w * 16 + r + (khalf ? 8 : 0);
            Sc[row * SEQ + kbase + nt * 16 + nlo] = acc[nt][r] * 0.125f;
        }
}

// ---------------------------------------------------------------------------
// Kernel 3: softmax(scores) @ V  -> out, 128 query rows per block
// ---------------------------------------------------------------------------
__global__ __launch_bounds__(256) void softmax_av_kernel(const float* __restrict__ Sc,
                                                         const unsigned short* __restrict__ Vb,
                                                         float* __restrict__ Out) {
    __shared__ __attribute__((aligned(16))) unsigned short Vt[64 * 32];  // [d][k] bf16, 4 KB
    __shared__ float redm[256];
    __shared__ float reds[256];
    __shared__ float rowMax[128];
    __shared__ float rowSumInv[128];

    const int t     = threadIdx.x;
    const int lane  = t & 31;
    const int w     = t >> 5;
    const int khalf = lane >> 4;
    const int nlo   = lane & 15;
    const int qbase = blockIdx.x * 128;
    const int b     = blockIdx.y;
    const float* ScB = Sc + (long)b * SEQ * SEQ;

    // ---- softmax stats: 2 threads per row, branchless two-pass scan ----
    {
        int row  = t >> 1;
        int half = t & 1;
        const float4* p4 = (const float4*)(ScB + (long)(qbase + row) * SEQ + half * (SEQ / 2));
        // pass 1: row max
        float m = -3.402823466e+38f;
        for (int i = 0; i < SEQ / 8; ++i) {          // 512 float4 loads
            float4 x = p4[i];
            m = fmaxf(m, fmaxf(fmaxf(x.x, x.y), fmaxf(x.z, x.w)));
        }
        // pass 2: sum of exp(x - m)
        float s = 0.0f;
        for (int i = 0; i < SEQ / 8; ++i) {
            float4 x = p4[i];
            s += __expf(x.x - m) + __expf(x.y - m) + __expf(x.z - m) + __expf(x.w - m);
        }
        redm[t] = m; reds[t] = s;
    }
    __syncthreads();
    if (t < 128) {
        float m0 = redm[2 * t], m1 = redm[2 * t + 1];
        float s0 = reds[2 * t], s1 = reds[2 * t + 1];
        float m  = fmaxf(m0, m1);
        float s  = s0 * __expf(m0 - m) + s1 * __expf(m1 - m);
        rowMax[t]    = m;
        rowSumInv[t] = 1.0f / s;
    }
    __syncthreads();

    // ---- weights @ V with WMMA, weights = exp(x - rowmax) built on the fly ----
    v8f acc[4];
    for (int nt = 0; nt < 4; ++nt)
        for (int i = 0; i < 8; ++i) acc[nt][i] = 0.0f;

    const int   mrow = qbase + w * 16 + nlo;     // A-matrix row for this lane
    const float mAx  = rowMax[w * 16 + nlo];
    const float* srow = ScB + (long)mrow * SEQ;
    unsigned int* VtW = (unsigned int*)Vt;

    for (int kt = 0; kt < SEQ / 32; ++kt) {
        const int k0 = kt * 32;
        // stage V chunk transposed & packed: VtW[d*16 + kk] = {V[k0+2kk][d], V[k0+2kk+1][d]}
#pragma unroll
        for (int i = 0; i < 4; ++i) {
            int e  = t + i * 256;                // 0..1023 dwords; e = kk*64 + d
            int d  = e & 63;
            int kk = e >> 6;                     // K-pair index 0..15
            unsigned int v0 = Vb[(long)(b * SEQ + k0 + 2 * kk) * DIMK + d];
            unsigned int v1 = Vb[(long)(b * SEQ + k0 + 2 * kk + 1) * DIMK + d];
            VtW[d * 16 + kk] = v0 | (v1 << 16);
        }
        __syncthreads();

        FragBF a;
#pragma unroll
        for (int j = 0; j < 8; ++j) {
            int koff = khalf * 16 + 2 * j;
            float2 xx = *(const float2*)(srow + k0 + koff);
            a.u[j] = pack_bf16(__expf(xx.x - mAx), __expf(xx.y - mAx));
        }
#pragma unroll
        for (int nt = 0; nt < 4; ++nt) {
            FragBF bb;
            const unsigned int* vr = VtW + (nt * 16 + nlo) * 16;  // 16 dwords/row
#pragma unroll
            for (int j = 0; j < 8; ++j) bb.u[j] = vr[khalf * 8 + j];
            acc[nt] = __builtin_amdgcn_wmma_f32_16x16x32_bf16(
                false, a.v, false, bb.v, (short)0, acc[nt], false, false);
        }
        __syncthreads();
    }

    // epilogue: normalize by row sum, store f32
#pragma unroll
    for (int nt = 0; nt < 4; ++nt)
#pragma unroll
        for (int r = 0; r < 8; ++r) {
            int rloc = w * 16 + r + (khalf ? 8 : 0);
            float invs = rowSumInv[rloc];
            Out[((long)b * SEQ + qbase + rloc) * DIMK + nt * 16 + nlo] = acc[nt][r] * invs;
        }
}

// ---------------------------------------------------------------------------
extern "C" void kernel_launch(void* const* d_in, const int* in_sizes, int n_in,
                              void* d_out, int out_size, void* d_ws, size_t ws_size,
                              hipStream_t stream) {
    const float* x_q = (const float*)d_in[0];
    const float* x_k = (const float*)d_in[1];
    const float* x_v = (const float*)d_in[2];
    const float* Wq  = (const float*)d_in[3];
    const float* bq  = (const float*)d_in[4];
    const float* Wk  = (const float*)d_in[5];
    const float* bk  = (const float*)d_in[6];
    const float* Wv  = (const float*)d_in[7];
    const float* bv  = (const float*)d_in[8];

    float* out    = (float*)d_out;                       // [B,S,64]
    float* scores = out + (long)NROWS * DIMK;            // [B,S,S]

    unsigned short* Qb = (unsigned short*)d_ws;          // bf16, 2 MB each
    unsigned short* Kb = Qb + (long)NROWS * DIMK;
    unsigned short* Vb = Kb + (long)NROWS * DIMK;

    dim3 blk(256);
    proj_kernel<<<NROWS / 128, blk, 0, stream>>>(x_q, Wq, bq, Qb);
    proj_kernel<<<NROWS / 128, blk, 0, stream>>>(x_k, Wk, bk, Kb);
    proj_kernel<<<NROWS / 128, blk, 0, stream>>>(x_v, Wv, bv, Vb);

    scores_kernel<<<dim3(SEQ / 128, SEQ / 128, BATCH), blk, 0, stream>>>(Qb, Kb, scores);

    softmax_av_kernel<<<dim3(SEQ / 128, BATCH), blk, 0, stream>>>(scores, Vb, out);
}
